// SparseMoE_69793218560576
// MI455X (gfx1250) — matrix operand order repaired and checked by reading, not compiled, not run on hardware
//
#include <hip/hip_runtime.h>

#define N_TOK 65536
#define DIN   512
#define DH    1024
#define DOUT  512
#define NEXP  8
#define ROWS  64          // tokens per MLP tile
#define MT    4           // ROWS/16 M-tiles per wave
#define NCHK  128         // columns per wave-pass half
#define THREADS 256       // 8 wave32
#define DHP   (DH + 8)    // padded LDS row stride (bf16 elems) -> conflict-free A reads

typedef __attribute__((ext_vector_type(16))) __bf16 v16bf;
typedef __attribute__((ext_vector_type(8)))  __bf16 v8bf;
typedef __attribute__((ext_vector_type(8)))  float  v8f;

union Frag  { v16bf v; v8bf h[2]; };
union Pack8 { v8bf v; __bf16 b[8]; unsigned u[4]; };

__device__ __forceinline__ __bf16 f2bf(float f) {
  union { float f; unsigned u; } in; in.f = f;
  unsigned r = in.u + 0x7FFFu + ((in.u >> 16) & 1u);   // round-to-nearest-even
  union { unsigned short s; __bf16 b; } out; out.s = (unsigned short)(r >> 16);
  return out.b;
}

__device__ __forceinline__ v8f wmma_bf16(v16bf a, v16bf b, v8f c) {
  return __builtin_amdgcn_wmma_f32_16x16x32_bf16(false, a, false, b, (short)0, c,
                                                 false, false);
}

// ---------------- gating: logits = x@Wg + bg, argmax ----------------
__global__ __launch_bounds__(THREADS)
void gate_kernel(const float* __restrict__ x, const float* __restrict__ Wg,
                 const float* __restrict__ bg, int* __restrict__ eid,
                 int* __restrict__ counts) {
  __shared__ float wg[NEXP][520];   // transposed [e][k], padded
  __shared__ float bgs[NEXP];
  const int tid = threadIdx.x;
  for (int i = tid; i < DIN * NEXP; i += THREADS) {
    int k = i / NEXP, e = i % NEXP;      // Wg row-major [DIN][E]; coalesced
    wg[e][k] = Wg[i];
  }
  if (tid < NEXP) bgs[tid] = bg[tid];
  __syncthreads();

  const int wave = tid >> 5, lane = tid & 31;
  const int n = blockIdx.x * 8 + wave;          // one wave per token
  float acc[NEXP];
#pragma unroll
  for (int e = 0; e < NEXP; ++e) acc[e] = 0.f;
  const float4* xr = (const float4*)(x + (size_t)n * DIN);
#pragma unroll
  for (int j = 0; j < 4; ++j) {
    const int k = j * 128 + lane * 4;
    float4 xv = xr[k >> 2];
#pragma unroll
    for (int e = 0; e < NEXP; ++e) {
      float4 wv = *(const float4*)&wg[e][k];
      acc[e] += xv.x * wv.x + xv.y * wv.y + xv.z * wv.z + xv.w * wv.w;
    }
  }
#pragma unroll
  for (int off = 16; off; off >>= 1)
#pragma unroll
    for (int e = 0; e < NEXP; ++e) acc[e] += __shfl_xor(acc[e], off, 32);
  if (lane == 0) {
    int best = 0; float bv = acc[0] + bgs[0];
#pragma unroll
    for (int e = 1; e < NEXP; ++e) {
      float v = acc[e] + bgs[e];
      if (v > bv) { bv = v; best = e; }        // first-max tie-break like jnp.argmax
    }
    eid[n] = best;
    atomicAdd(&counts[best], 1);
  }
}

// ---------------- tiny routing kernels ----------------
__global__ void zero_kernel(int* ws) { if (threadIdx.x < 16) ws[threadIdx.x] = 0; }

__global__ void scan_kernel(int* ws) {  // ws: counts[0..7] cursors[8..15] offsets[16..24]
  if (threadIdx.x == 0) {
    int s = 0;
    for (int e = 0; e < NEXP; ++e) { ws[16 + e] = s; s += ws[e]; ws[8 + e] = 0; }
    ws[16 + NEXP] = s;
  }
}

__global__ __launch_bounds__(THREADS)
void scatter_kernel(const int* __restrict__ eid, const int* __restrict__ offsets,
                    int* __restrict__ cursors, int* __restrict__ perm) {
  const int n = blockIdx.x * THREADS + threadIdx.x;
  const int e = eid[n];
  const int slot = atomicAdd(&cursors[e], 1);
  perm[offsets[e] + slot] = n;
}

// -------- weights fp32 -> bf16, swizzled into per-lane WMMA B-fragment order -----
// dst layout per expert: frag(nb, kb) at ((nb*kSteps + kb)*32 + lane)*16 + i
//   lane 0-15 : column nb*16 + lane,      K = kb*32 + i        (i = 0..15)
//   lane 16-31: column nb*16 + (lane-16), K = kb*32 + 16 + i
__global__ __launch_bounds__(THREADS)
void wswz_kernel(const float* __restrict__ src, __bf16* __restrict__ dst,
                 int K, int Nout) {
  __shared__ float tile[32][132];
  const int kSteps = K >> 5;
  const int blocksPerExpert = kSteps * (Nout >> 7);
  const int b = blockIdx.x;
  const int e = b / blocksPerExpert;
  const int r = b % blocksPerExpert;
  const int kblk = r % kSteps;
  const int nb0 = (r / kSteps) * 8;          // 8 col-blocks = 128 cols per block
  const size_t S = (size_t)K * Nout;
  const float* sp = src + (size_t)e * S;
  __bf16* dp = dst + (size_t)e * S;
  const int tid = threadIdx.x;
  for (int i = tid; i < 32 * 128; i += THREADS) {
    const int kk = i >> 7, c = i & 127;      // coalesced f32 loads
    tile[kk][c] = sp[(size_t)(kblk * 32 + kk) * Nout + nb0 * 16 + c];
  }
  __syncthreads();
  for (int i = tid; i < 32 * 128; i += THREADS) {
    const int nbl = i >> 9;                  // 0..7
    const int rr = i & 511;
    const int lane = rr >> 4, ii = rr & 15;  // consecutive tid -> consecutive dst
    const float v = tile[(lane >> 4) * 16 + ii][nbl * 16 + (lane & 15)];
    dp[((size_t)((nb0 + nbl) * kSteps + kblk) * 32 + lane) * 16 + ii] = f2bf(v);
  }
}

// ---------------- one dense layer over a 64-token LDS tile ----------------
template <int K, int Nout, bool FINAL>
__device__ __forceinline__ void run_layer(
    const __bf16* __restrict__ actIn, __bf16* __restrict__ actOut,
    const __bf16* __restrict__ Wf,    // swizzled bf16 fragments (see wswz_kernel)
    const float* __restrict__ bias,
    float* __restrict__ outG, const int* __restrict__ permS, int tid)
{
  const int wave = tid >> 5, lane = tid & 31;
  const int laneN = lane & 15;
  const int hiK = lane >> 4;                 // lanes 16-31 own the upper K half
  constexpr int kSteps = K >> 5;

  int tok[MT][8];                            // final layer: per-lane output row tokens
  if constexpr (FINAL) {
#pragma unroll
    for (int m = 0; m < MT; ++m) {
      const int rbase = m * 16 + hiK * 8;
#pragma unroll
      for (int rr = 0; rr < 8; ++rr) tok[m][rr] = permS[rbase + rr];
    }
  }

#pragma unroll 1
  for (int nc = 0; nc < Nout; nc += 2 * NCHK) {
    const int nA = nc + wave * 16 + laneN;
    const float bv0 = bias[nA];
    const float bv1 = bias[nA + NCHK];
    v8f acc[2][MT];
#pragma unroll
    for (int s = 0; s < 2; ++s)
#pragma unroll
      for (int m = 0; m < MT; ++m)
        acc[s][m] = (v8f){0.f, 0.f, 0.f, 0.f, 0.f, 0.f, 0.f, 0.f};
    const int nb0 = (nc >> 4) + wave;
    const __bf16* wp0 = Wf + ((size_t)nb0 * kSteps * 32 + lane) * 16;
    const __bf16* wp1 = Wf + ((size_t)(nb0 + 8) * kSteps * 32 + lane) * 16;
#pragma unroll 1
    for (int k0 = 0; k0 < K; k0 += 32) {
      Frag b0f, b1f;
      b0f.h[0] = *(const v8bf*)wp0; b0f.h[1] = *(const v8bf*)(wp0 + 8);
      b1f.h[0] = *(const v8bf*)wp1; b1f.h[1] = *(const v8bf*)(wp1 + 8);
      if (k0 + 32 < K) {
        __builtin_prefetch(wp0 + 512, 0, 1);
        __builtin_prefetch(wp1 + 512, 0, 1);
      }
      wp0 += 512; wp1 += 512;                // next K-step fragment (1 KB / wave)
#pragma unroll
      for (int m = 0; m < MT; ++m) {
        const __bf16* ap = actIn + (size_t)(m * 16 + laneN) * DHP + k0 + hiK * 16;
        Frag a;
        a.h[0] = *(const v8bf*)ap;           // K = k0+16*hiK .. +7
        a.h[1] = *(const v8bf*)(ap + 8);     // K = k0+16*hiK+8 .. +15
        acc[0][m] = wmma_bf16(a.v, b0f.v, acc[0][m]);   // A reused by both strips
        acc[1][m] = wmma_bf16(a.v, b1f.v, acc[1][m]);
      }
    }
    // epilogue: bias (+ReLU -> LDS bf16) or final scatter-store to global fp32
#pragma unroll
    for (int s = 0; s < 2; ++s) {
      const int n = nA + s * NCHK;
      const float bv = s ? bv1 : bv0;
#pragma unroll
      for (int m = 0; m < MT; ++m) {
#pragma unroll
        for (int rr = 0; rr < 8; ++rr) {
          const float v = acc[s][m][rr] + bv;
          if constexpr (FINAL) {
            const int t = tok[m][rr];
            if (t >= 0) outG[(size_t)t * DOUT + n] = v;
          } else {
            const int row = m * 16 + rr + hiK * 8;   // C/D layout: VGPR rr -> M (+8 hi)
            actOut[(size_t)row * DHP + n] = f2bf(v > 0.f ? v : 0.f);
          }
        }
      }
    }
  }
}

// ---------------- fused 3-layer MLP over gathered 64-token tiles ----------------
__global__ __launch_bounds__(THREADS, 1)
void moe_mlp_kernel(const float* __restrict__ x,
                    const __bf16* __restrict__ W0f, const float* __restrict__ b0,
                    const __bf16* __restrict__ W1f, const float* __restrict__ b1,
                    const __bf16* __restrict__ W2f, const float* __restrict__ b2,
                    const int* __restrict__ offsets, const int* __restrict__ perm,
                    float* __restrict__ out)
{
  __shared__ __bf16 actA[ROWS * DHP];   // 129 KB
  __shared__ __bf16 actB[ROWS * DHP];   // 129 KB
  __shared__ int    permS[ROWS];

  const int tid = threadIdx.x;
  // locate (expert, tile) from device-side segment sizes
  int e = -1, tile = 0, cum = 0;
  for (int i = 0; i < NEXP; ++i) {
    const int c = offsets[i + 1] - offsets[i];
    const int t = (c + ROWS - 1) / ROWS;
    if (e < 0 && (int)blockIdx.x < cum + t) { e = i; tile = (int)blockIdx.x - cum; }
    cum += t;
  }
  if (e < 0) return;
  const int base  = offsets[e] + tile * ROWS;
  int valid = offsets[e + 1] - base;
  if (valid > ROWS) valid = ROWS;

  if (tid < ROWS) permS[tid] = (tid < valid) ? perm[base + tid] : -1;
  __syncthreads();
  // gather x rows -> bf16 tile: 2x float4 global loads -> 1x ds_store_b128
  for (int i = tid; i < ROWS * (DIN / 8); i += THREADS) {
    const int r = i / (DIN / 8), kc = (i % (DIN / 8)) * 8;
    const int t = permS[r];
    Pack8 p;
    if (t >= 0) {
      const float4* xp = (const float4*)(x + (size_t)t * DIN + kc);
      const float4 a = xp[0], b = xp[1];
      p.b[0] = f2bf(a.x); p.b[1] = f2bf(a.y); p.b[2] = f2bf(a.z); p.b[3] = f2bf(a.w);
      p.b[4] = f2bf(b.x); p.b[5] = f2bf(b.y); p.b[6] = f2bf(b.z); p.b[7] = f2bf(b.w);
    } else {
      p.u[0] = 0u; p.u[1] = 0u; p.u[2] = 0u; p.u[3] = 0u;
    }
    *(v8bf*)(actA + (size_t)r * DHP + kc) = p.v;
  }
  __syncthreads();
  run_layer<DIN, DH, false>(actA, actB, W0f + (size_t)e * DIN * DH, b0 + e * DH,
                            nullptr, permS, tid);
  __syncthreads();
  run_layer<DH, DH, false>(actB, actA, W1f + (size_t)e * DH * DH, b1 + e * DH,
                           nullptr, permS, tid);
  __syncthreads();
  run_layer<DH, DOUT, true>(actA, nullptr, W2f + (size_t)e * DH * DOUT,
                            b2 + e * DOUT, out, permS, tid);
}

// ---------------- host launcher ----------------
extern "C" void kernel_launch(void* const* d_in, const int* in_sizes, int n_in,
                              void* d_out, int out_size, void* d_ws, size_t ws_size,
                              hipStream_t stream) {
  const float* x  = (const float*)d_in[0];
  const float* Wg = (const float*)d_in[1];
  const float* bg = (const float*)d_in[2];
  const float* W0 = (const float*)d_in[3];
  const float* b0 = (const float*)d_in[4];
  const float* W1 = (const float*)d_in[5];
  const float* b1 = (const float*)d_in[6];
  const float* W2 = (const float*)d_in[7];
  const float* b2 = (const float*)d_in[8];
  float* out = (float*)d_out;

  int* wsI     = (int*)d_ws;
  int* offsets = wsI + 16;       // [16..24]
  int* cursors = wsI + 8;        // [8..15]
  int* eid     = wsI + 32;       // N ints
  int* perm    = eid + N_TOK;    // N ints
  __bf16* wbf  = (__bf16*)((char*)d_ws + (1 << 20));   // swizzled bf16 weights (~32 MB)
  __bf16* W0f  = wbf;
  __bf16* W1f  = W0f + (size_t)NEXP * DIN * DH;
  __bf16* W2f  = W1f + (size_t)NEXP * DH  * DH;
  (void)in_sizes; (void)n_in; (void)out_size; (void)ws_size;

  zero_kernel   <<<1, 32, 0, stream>>>(wsI);
  gate_kernel   <<<N_TOK / 8, THREADS, 0, stream>>>(x, Wg, bg, eid, wsI);
  scan_kernel   <<<1, 32, 0, stream>>>(wsI);
  scatter_kernel<<<N_TOK / THREADS, THREADS, 0, stream>>>(eid, offsets, cursors, perm);
  wswz_kernel   <<<NEXP * (DIN / 32) * (DH   / 128), THREADS, 0, stream>>>(W0, W0f, DIN, DH);
  wswz_kernel   <<<NEXP * (DH  / 32) * (DH   / 128), THREADS, 0, stream>>>(W1, W1f, DH,  DH);
  wswz_kernel   <<<NEXP * (DH  / 32) * (DOUT / 128), THREADS, 0, stream>>>(W2, W2f, DH,  DOUT);
  moe_mlp_kernel<<<N_TOK / ROWS + NEXP, THREADS, 0, stream>>>(
      x, W0f, b0, W1f, b1, W2f, b2, offsets, perm, out);
}
